// AMIPRouterInference_51522427683261
// MI455X (gfx1250) — compile-verified, matrix operand order checked by reading, not run on hardware
//
#include <hip/hip_runtime.h>
#include <stdint.h>

// ---------------- problem constants ----------------
#define Bb   2
#define Ll   2048
#define Dd   1024
#define Kk   8
#define D4   256
#define NNB  10            // 2*R neighbors
#define NT   (Bb * Ll)     // 4096 tokens
#define J1   4096          // GEMM1 N  = 2*K*D4
#define K2T  2080          // GEMM2 K  = K*D4 + 32 (bias columns, padded)

typedef unsigned short u16;
typedef __bf16 bf16_t;
typedef bf16_t v16bf __attribute__((ext_vector_type(16)));
typedef bf16_t v8bf  __attribute__((ext_vector_type(8)));
typedef float  v8f   __attribute__((ext_vector_type(8)));

__device__ __forceinline__ u16 f2bf(float x) {
    union { float f; uint32_t u; } c; c.f = x;
    uint32_t u = c.u;
    uint32_t r = u + 0x7FFFu + ((u >> 16) & 1u);           // RNE
    if ((u & 0x7F800000u) == 0x7F800000u) r = u;            // inf/nan passthrough
    return (u16)(r >> 16);
}

__device__ __forceinline__ float gelu_exact(float x) {
    return 0.5f * x * (1.0f + erff(x * 0.70710678118654752f));
}

__device__ __forceinline__ int neigh_off(int n) { return (n < 5) ? (n - 5) : (n - 4); }

// ---- CDNA5 async global->LDS (ASYNCcnt path), inline asm: GVS addressing ----
__device__ __forceinline__ void async_ld_b128(const void* sbase, uint32_t goff_bytes,
                                              void* lds_ptr) {
    uint32_t laddr = (uint32_t)(uintptr_t)lds_ptr;   // LDS aperture: low 32 bits = LDS offset
    asm volatile("global_load_async_to_lds_b128 %0, %1, %2"
                 :: "v"(laddr), "v"(goff_bytes), "s"(sbase)
                 : "memory");
}
__device__ __forceinline__ void wait_async0() {
    asm volatile("s_wait_asynccnt 0x0" ::: "memory");
}

// ---------------- 1) fp32 -> bf16 cast of h_L ----------------
__global__ __launch_bounds__(256) void k_cast_h(const float* __restrict__ h, u16* __restrict__ hb) {
    int i = blockIdx.x * 256 + threadIdx.x;       // NT*D total
    hb[i] = f2bf(h[i]);
}

// ---------------- 2) W1 (K,2D,D4) -> w1t[j][d] bf16, j = (half,k,f) ----------------
__global__ __launch_bounds__(256) void k_prep_w1(const float* __restrict__ W1, u16* __restrict__ w1t) {
    int id = blockIdx.x * 256 + threadIdx.x;      // J1*D total
    int j = id >> 10;                             // 0..4095
    int d = id & 1023;
    int k = (j & 2047) >> 8;
    int f = j & 255;
    int dd = (j < 2048) ? d : (d + Dd);           // first half: anchor rows, second: self rows
    w1t[(size_t)j * Dd + d] = f2bf(W1[((size_t)k * (2 * Dd) + dd) * D4 + f]);
}

// ---------------- 3) W2 (K,D4,D)+b2 -> w2t[d][p] bf16, p in [0,2080) ----------------
__global__ __launch_bounds__(256) void k_prep_w2(const float* __restrict__ W2,
                                                 const float* __restrict__ b2,
                                                 u16* __restrict__ w2t) {
    int d = blockIdx.x;                           // 1024 blocks
    u16* row = w2t + (size_t)d * K2T;
    for (int p = threadIdx.x; p < K2T; p += 256) {
        float v;
        if (p < 2048) {
            int k = p >> 8, f = p & 255;
            v = W2[((size_t)k * D4 + f) * Dd + d];
        } else {
            int k2 = p - 2048;
            v = (k2 < Kk) ? b2[(size_t)k2 * Dd + d] : 0.0f;
        }
        row[p] = f2bf(v);
    }
}

// ---------------- 4) router softmax + neighbor-score softmax ----------------
__global__ __launch_bounds__(256) void k_router(const float* __restrict__ hL,
                                                const int*   __restrict__ mask,
                                                const float* __restrict__ Wr,
                                                const float* __restrict__ br,
                                                float* __restrict__ wout,    // [NT][8]
                                                float* __restrict__ cwout,   // [NT][10]
                                                float* __restrict__ haout) { // [NT]
    const int wid  = (blockIdx.x << 3) + (threadIdx.x >> 5);  // one wave32 per token
    const int lane = threadIdx.x & 31;
    const int b = wid >> 11, l = wid & (Ll - 1);
    const float* h = hL + (size_t)wid * Dd;

    float hreg[32];
    #pragma unroll
    for (int i = 0; i < 32; ++i) hreg[i] = h[lane + (i << 5)];

    // router logits
    float lg[Kk];
    #pragma unroll
    for (int k = 0; k < Kk; ++k) lg[k] = 0.0f;
    #pragma unroll 4
    for (int i = 0; i < 32; ++i) {
        const float* wr = Wr + (size_t)(lane + (i << 5)) * Kk;
        #pragma unroll
        for (int k = 0; k < Kk; ++k) lg[k] = fmaf(hreg[i], wr[k], lg[k]);
    }
    #pragma unroll
    for (int m = 16; m > 0; m >>= 1)
        #pragma unroll
        for (int k = 0; k < Kk; ++k) lg[k] += __shfl_xor(lg[k], m, 32);
    #pragma unroll
    for (int k = 0; k < Kk; ++k) lg[k] += br[k];

    float mx = lg[0];
    #pragma unroll
    for (int k = 1; k < Kk; ++k) mx = fmaxf(mx, lg[k]);
    float wv[Kk], se = 0.0f;
    #pragma unroll
    for (int k = 0; k < Kk; ++k) { wv[k] = __expf(lg[k] - mx); se += wv[k]; }
    float inv = 1.0f / se;
    #pragma unroll
    for (int k = 0; k < Kk; ++k) wv[k] *= inv;

    // neighbor scores
    const bool selfmask = (mask[wid] == 1);
    float sc[NNB]; bool val[NNB];
    #pragma unroll
    for (int n = 0; n < NNB; ++n) {
        int off = neigh_off(n);
        int nr  = l + off;
        bool inb = (nr >= 0) && (nr < Ll);
        int nc  = nr < 0 ? 0 : (nr > Ll - 1 ? Ll - 1 : nr);
        int ntok = (b << 11) + nc;
        val[n] = inb && selfmask && (mask[ntok] != 1);
        const float* ha = hL + (size_t)ntok * Dd;
        float dt = 0.0f;
        #pragma unroll 4
        for (int i = 0; i < 32; ++i) dt = fmaf(hreg[i], ha[lane + (i << 5)], dt);
        #pragma unroll
        for (int m = 16; m > 0; m >>= 1) dt += __shfl_xor(dt, m, 32);
        sc[n] = dt * 0.03125f;                    // 1/sqrt(1024)
    }
    float mxs = -1e30f; bool any = false;
    #pragma unroll
    for (int n = 0; n < NNB; ++n) if (val[n]) { mxs = fmaxf(mxs, sc[n]); any = true; }
    float e[NNB], den = 0.0f;
    #pragma unroll
    for (int n = 0; n < NNB; ++n) { e[n] = val[n] ? __expf(sc[n] - mxs) : 0.0f; den += e[n]; }
    den = fmaxf(den, 1e-30f);
    inv = 1.0f / den;

    if (lane == 0) {
        #pragma unroll
        for (int k = 0; k < Kk; ++k) wout[(size_t)wid * Kk + k] = wv[k];
        #pragma unroll
        for (int n = 0; n < NNB; ++n) cwout[(size_t)wid * NNB + n] = e[n] * inv;
        haout[wid] = any ? 1.0f : 0.0f;
    }
}

// ---------------- WMMA bf16 GEMM: C[M][N] = A[M][Kt] * Bt[N][Kt]^T ----------------
// A, Bt row-major bf16 with leading dim Kt. C fp32 with leading dim ldc.
// Block: 256 threads (8 waves), 128x128 tile, BK=32. Wave owns 64x32 (4x2 WMMA tiles).
// Double-buffered LDS staged with global_load_async_to_lds_b128 (ASYNCcnt).
#define BM 128
#define BN 128
#define BK 32
#define LPAD 8     // row stride 40 u16 = 80B (16B aligned)

__global__ __launch_bounds__(256) void k_gemm_bf16(const u16* __restrict__ A,
                                                   const u16* __restrict__ Bt,
                                                   float* __restrict__ C,
                                                   int Kt, int ldc) {
    __shared__ u16 As[2][BM][BK + LPAD];
    __shared__ u16 Bs[2][BN][BK + LPAD];

    const int tid  = threadIdx.x;
    const int lane = tid & 31;
    const int wave = tid >> 5;
    const int m0 = blockIdx.y * BM;
    const int n0 = blockIdx.x * BN;
    const int wm = (wave & 1) * 64;     // wave row offset in tile
    const int wn = (wave >> 1) * 32;    // wave col offset in tile

    const int lr = tid >> 1;            // 0..127 : staging row
    const int lc = (tid & 1) * 16;      // 0 or 16: staging col
    const int frow = lane & 15;         // fragment row within 16
    const int fcol = (lane >> 4) * 8;   // fragment K sub-offset (0 or 8)
    const int hi8  = (lane >> 4) * 8;   // C-row offset for upper half-wave

    const uint32_t ga_row = (uint32_t)(m0 + lr) * (uint32_t)Kt + lc;  // element offsets
    const uint32_t gb_row = (uint32_t)(n0 + lr) * (uint32_t)Kt + lc;

    const v8f vzero = {0.f,0.f,0.f,0.f,0.f,0.f,0.f,0.f};
    v8f acc[4][2];
    #pragma unroll
    for (int i = 0; i < 4; ++i)
        #pragma unroll
        for (int j = 0; j < 2; ++j) acc[i][j] = vzero;

    // stage tile k0 into buffer buf (4 async b128 per thread, no VGPR bounce)
    auto stage = [&](int buf, int k0) {
        uint32_t ao = (ga_row + (uint32_t)k0) * 2u;   // bytes
        uint32_t bo = (gb_row + (uint32_t)k0) * 2u;
        async_ld_b128(A,  ao,       &As[buf][lr][lc]);
        async_ld_b128(A,  ao + 16u, &As[buf][lr][lc + 8]);
        async_ld_b128(Bt, bo,       &Bs[buf][lr][lc]);
        async_ld_b128(Bt, bo + 16u, &Bs[buf][lr][lc + 8]);
    };

    stage(0, 0);
    wait_async0();
    __syncthreads();

    int buf = 0;
    for (int k0 = 0; k0 < Kt; k0 += BK) {
        const int nk = k0 + BK;
        if (nk < Kt) stage(buf ^ 1, nk);     // prefetch next tile while computing

        v16bf af[4], bfr[2];
        #pragma unroll
        for (int mt = 0; mt < 4; ++mt) {
            const int r = wm + mt * 16 + frow;
            v8bf lo = *reinterpret_cast<const v8bf*>(&As[buf][r][fcol]);
            v8bf hi = *reinterpret_cast<const v8bf*>(&As[buf][r][fcol + 16]);
            af[mt] = __builtin_shufflevector(lo, hi, 0,1,2,3,4,5,6,7,8,9,10,11,12,13,14,15);
        }
        #pragma unroll
        for (int nt = 0; nt < 2; ++nt) {
            const int r = wn + nt * 16 + frow;
            v8bf lo = *reinterpret_cast<const v8bf*>(&Bs[buf][r][fcol]);
            v8bf hi = *reinterpret_cast<const v8bf*>(&Bs[buf][r][fcol + 16]);
            bfr[nt] = __builtin_shufflevector(lo, hi, 0,1,2,3,4,5,6,7,8,9,10,11,12,13,14,15);
        }
        #pragma unroll
        for (int mt = 0; mt < 4; ++mt)
            #pragma unroll
            for (int nt = 0; nt < 2; ++nt)
                acc[mt][nt] = __builtin_amdgcn_wmma_f32_16x16x32_bf16(
                    false, af[mt], false, bfr[nt], (short)0, acc[mt][nt], false, false);

        if (nk < Kt) wait_async0();          // own async stores to LDS landed
        __syncthreads();                      // everyone done reading buf / staging buf^1
        buf ^= 1;
    }

    // epilogue: lane holds (M = r + hi8, N = lane&15) for VGPR r of each tile
    #pragma unroll
    for (int mt = 0; mt < 4; ++mt) {
        #pragma unroll
        for (int nt = 0; nt < 2; ++nt) {
            const int mb = m0 + wm + mt * 16 + hi8;
            const int nb = n0 + wn + nt * 16 + (lane & 15);
            #pragma unroll
            for (int r = 0; r < 8; ++r)
                C[(size_t)(mb + r) * ldc + nb] = acc[mt][nt][r];
        }
    }
}

// ---------------- 6) gelu + neighbor combine -> g (bf16, [NT][2080]) ----------------
__global__ __launch_bounds__(256) void k_gelu(const float* __restrict__ U,     // [NT][4096]
                                              const float* __restrict__ b1,    // [2048]
                                              const float* __restrict__ wbuf,  // [NT][8]
                                              const float* __restrict__ cwbuf, // [NT][10]
                                              const float* __restrict__ habuf, // [NT]
                                              u16* __restrict__ g) {           // [NT][2080]
    const int t = blockIdx.x;
    const int b = t >> 11, l = t & (Ll - 1);

    float cw[NNB];
    #pragma unroll
    for (int n = 0; n < NNB; ++n) cw[n] = cwbuf[(size_t)t * NNB + n];
    float wv[Kk];
    #pragma unroll
    for (int k = 0; k < Kk; ++k) wv[k] = wbuf[(size_t)t * Kk + k];
    const float ha = habuf[t];

    const float* Un[NNB];
    #pragma unroll
    for (int n = 0; n < NNB; ++n) {
        int nr = l + neigh_off(n);
        int nc = nr < 0 ? 0 : (nr > Ll - 1 ? Ll - 1 : nr);
        Un[n] = U + (size_t)((b << 11) + nc) * J1;          // anchor half: cols [0,2048)
    }
    const float* Us = U + (size_t)t * J1 + 2048;            // self half
    u16* gr = g + (size_t)t * K2T;

    for (int p = threadIdx.x; p < 2048; p += 256) {
        const float base = Us[p] + b1[p];
        float a = 0.0f;
        #pragma unroll
        for (int n = 0; n < NNB; ++n)
            a = fmaf(cw[n], gelu_exact(Un[n][p] + base), a);
        gr[p] = f2bf(wv[p >> 8] * a);
    }
    if (threadIdx.x < 32) {                                  // folded w*b2 bias columns
        float v = (threadIdx.x < Kk) ? ha * wv[threadIdx.x] : 0.0f;
        gr[2048 + threadIdx.x] = f2bf(v);
    }
}

// ---------------- launcher ----------------
extern "C" void kernel_launch(void* const* d_in, const int* in_sizes, int n_in,
                              void* d_out, int out_size, void* d_ws, size_t ws_size,
                              hipStream_t stream) {
    (void)in_sizes; (void)n_in; (void)out_size; (void)ws_size;

    const float* h_L  = (const float*)d_in[0];
    const int*   mask = (const int*)  d_in[1];
    const float* Wr   = (const float*)d_in[2];
    const float* br   = (const float*)d_in[3];
    const float* W1   = (const float*)d_in[4];
    const float* b1   = (const float*)d_in[5];
    const float* W2   = (const float*)d_in[6];
    const float* b2   = (const float*)d_in[7];
    float* out = (float*)d_out;

    char* ws = (char*)d_ws;
    size_t off = 0;
    auto alloc = [&](size_t bytes) -> void* {
        void* p = ws + off;
        off += (bytes + 255) & ~(size_t)255;
        return p;
    };
    u16*   hbf  = (u16*)  alloc((size_t)NT * Dd * 2);        // bf16 h_L
    u16*   w1t  = (u16*)  alloc((size_t)J1 * Dd * 2);        // repacked W1 (N-major)
    u16*   w2t  = (u16*)  alloc((size_t)Dd * K2T * 2);       // repacked W2+b2
    float* U    = (float*)alloc((size_t)NT * J1 * 4);        // GEMM1 out
    u16*   g    = (u16*)  alloc((size_t)NT * K2T * 2);       // gated gelu, bf16
    float* wbuf = (float*)alloc((size_t)NT * Kk * 4);
    float* cwb  = (float*)alloc((size_t)NT * NNB * 4);
    float* hab  = (float*)alloc((size_t)NT * 4);

    k_cast_h <<<(NT * Dd) / 256, 256, 0, stream>>>(h_L, hbf);
    k_prep_w1<<<(J1 * Dd) / 256, 256, 0, stream>>>(W1, w1t);
    k_prep_w2<<<Dd, 256, 0, stream>>>(W2, b2, w2t);
    k_router <<<NT / 8, 256, 0, stream>>>(h_L, mask, Wr, br, wbuf, cwb, hab);

    // GEMM1: U[4096][4096] = hbf[4096][1024] x w1t[4096][1024]^T
    k_gemm_bf16<<<dim3(J1 / BN, NT / BM), 256, 0, stream>>>(hbf, w1t, U, Dd, J1);

    k_gelu<<<NT, 256, 0, stream>>>(U, b1, wbuf, cwb, hab, g);

    // GEMM2: out[4096][1024] = g[4096][2080] x w2t[1024][2080]^T
    k_gemm_bf16<<<dim3(Dd / BN, NT / BM), 256, 0, stream>>>(g, w2t, out, K2T, Dd);
}
